// TransformerSentenceEncoderLayer_29609504539206
// MI455X (gfx1250) — matrix operand order repaired and checked by reading, not compiled, not run on hardware
//
#include <hip/hip_runtime.h>

typedef __attribute__((ext_vector_type(16))) _Float16 v16h;
typedef __attribute__((ext_vector_type(8)))  float    v8f;
typedef __attribute__((ext_vector_type(4)))  int      v4i;

#define kT   2048
#define kB   2
#define kD   768
#define kH   8
#define kHD  96
#define kFFN 3072
#define kN   (kT * kB)   // 4096 tokens

// CDNA5 async global->LDS path (ASYNCcnt-tracked), with safe fallback
#if defined(__HIP_DEVICE_COMPILE__) && __has_builtin(__builtin_amdgcn_global_load_async_to_lds_b128)
#define HAVE_ASYNC_LDS 1
#else
#define HAVE_ASYNC_LDS 0
#endif

#if HAVE_ASYNC_LDS
__device__ __forceinline__ void async_cp16(_Float16* lds_dst, const _Float16* gsrc) {
  // signature: (v4i addrspace(1)* src, v4i addrspace(3)* dst, imm offset, imm cpol)
  __builtin_amdgcn_global_load_async_to_lds_b128(
      (__attribute__((address_space(1))) v4i*)gsrc,
      (__attribute__((address_space(3))) v4i*)lds_dst,
      0, 0);
}
#endif

template <int N>
__device__ __forceinline__ void wait_async() {
#if defined(__HIP_DEVICE_COMPILE__)
#if __has_builtin(__builtin_amdgcn_s_wait_asynccnt)
  __builtin_amdgcn_s_wait_asynccnt((unsigned short)N);
#else
  asm volatile("s_wait_asynccnt %0" ::"i"(N) : "memory");
#endif
#endif
}

// ---------------------------------------------------------------------------
// WMMA helpers (CDNA5 16x16x32 f16 -> f32)
// ---------------------------------------------------------------------------
__device__ __forceinline__ v8f wmma_f16(v16h a, v16h b, v8f c) {
  return __builtin_amdgcn_wmma_f32_16x16x32_f16(
      /*neg_a=*/false, a, /*neg_b=*/false, b,
      /*c_mod=*/(short)0, c, /*reuse_a=*/false, /*reuse_b=*/false);
}

// A fragment: 16x32 (MxK) f16. Order used here: lane m = lane&15, K-half by lane[4];
// 16 contiguous k per lane: k = kh + 2v, packed pairs.
__device__ __forceinline__ v16h frag_a(const _Float16* A, int lda, int row0, int k0) {
  const int lane = threadIdx.x & 31;
  const int m  = lane & 15;
  const int kh = (lane >> 4) << 4;
  const _Float16* p = A + (long)(row0 + m) * lda + k0 + kh;
  union { unsigned u[8]; v16h v; } r;
  const unsigned* pa = (const unsigned*)p;        // k = kh + {0,2,4,6}
  const unsigned* pb = (const unsigned*)(p + 8);  // k = kh + 8 + {0,2,4,6}
#pragma unroll
  for (int i = 0; i < 4; ++i) { r.u[i] = pa[i]; r.u[4 + i] = pb[i]; }
  return r.v;
}

// B fragment for C = A @ W^T: B(k,n) = W[n, k], W row-major [J x K].
// Order used here (as frag_a): lane n = lane&15, K-half by lane[4]; 16 contiguous k per lane.
__device__ __forceinline__ v16h frag_b_wt(const _Float16* W, int ldw, int j0, int k0) {
  const int lane = threadIdx.x & 31;
  const _Float16* p = W + (long)(j0 + (lane & 15)) * ldw + k0 + ((lane >> 4) << 4);
  union { unsigned u[8]; v16h v; } r;
  const unsigned* q = (const unsigned*)p;
#pragma unroll
  for (int i = 0; i < 8; ++i) r.u[i] = q[i];
  return r.v;
}

// ---------------------------------------------------------------------------
// fp32 -> fp16 convert (grid-stride)
// ---------------------------------------------------------------------------
__global__ __launch_bounds__(256) void f32_to_f16_k(const float* __restrict__ in,
                                                    _Float16* __restrict__ out, long n) {
  long i = (long)blockIdx.x * blockDim.x + threadIdx.x;
  long stride = (long)gridDim.x * blockDim.x;
  for (; i < n; i += stride) out[i] = (_Float16)in[i];
}

// ---------------------------------------------------------------------------
// GEMM: out[n,j] = sum_k A[n,k] * W[j,k]  (+bias)(*scl)(relu)(+resid)
// Block = 8 waves stacked on M (256 rows) sharing one 64-col B tile that is
// double-buffered in LDS via async global->LDS copies (ASYNCcnt). Each wave
// computes 32x64 (2 A-frags x 4 B-frags -> 8 WMMA per K=32 step).
// LDS B row stride = 40 halves: 16B-aligned rows for b128 async writes and
// conflict-free (20*n mod 64 distinct) fragment reads.
// ---------------------------------------------------------------------------
#define BSTRIDE 40
__global__ __launch_bounds__(256) void gemm_wt_k(
    const _Float16* __restrict__ A, const _Float16* __restrict__ W,
    const float* __restrict__ bias, const float* __restrict__ resid,
    float* __restrict__ outF, _Float16* __restrict__ outH, _Float16* __restrict__ outQKV,
    int N, int J, int K, float scl, int doRelu) {
  __shared__ _Float16 ldsB[2][64 * BSTRIDE];
  const int tid  = threadIdx.x;
  const int wave = tid >> 5;
  const int lane = tid & 31;
  const int jb   = J >> 6;
  const int row0 = ((blockIdx.x / jb) << 8) + (wave << 5);  // wave's 32 rows
  const int col0 = (blockIdx.x % jb) << 6;

  // staging: 256 threads x 16B = one 64x32 f16 tile per K-step
  const int sj = tid >> 2;            // 0..63  (column j within tile)
  const int sp = (tid & 3) << 3;      // 0,8,16,24 (k offset in halves)
  const _Float16* Wrow = W + (long)(col0 + sj) * K + sp;
  _Float16* lrow[2] = { &ldsB[0][sj * BSTRIDE + sp], &ldsB[1][sj * BSTRIDE + sp] };

  auto stage = [&](int buf, int k0) {
#if HAVE_ASYNC_LDS
    async_cp16(lrow[buf], Wrow + k0);
#else
    uint4 t = *(const uint4*)(Wrow + k0);
    *(uint4*)(lrow[buf]) = t;
#endif
  };

  v8f c[2][4];
  v8f zero = {};
#pragma unroll
  for (int mi = 0; mi < 2; ++mi)
#pragma unroll
    for (int t = 0; t < 4; ++t) c[mi][t] = zero;

  stage(0, 0);
  int buf = 0;
  for (int k0 = 0; k0 < K; k0 += 32, buf ^= 1) {
#if HAVE_ASYNC_LDS
    wait_async<0>();   // this wave's copy into ldsB[buf] has landed
#endif
    __syncthreads();   // everyone's copies landed; prev readers of buf^1 done
    if (k0 + 32 < K) stage(buf ^ 1, k0 + 32);

    v16h a0 = frag_a(A, K, row0,      k0);
    v16h a1 = frag_a(A, K, row0 + 16, k0);
    const _Float16* Bb = ldsB[buf];
#pragma unroll
    for (int t = 0; t < 4; ++t) {
      v16h b = frag_b_wt(Bb, BSTRIDE, t * 16, 0);
      c[0][t] = wmma_f16(a0, b, c[0][t]);
      c[1][t] = wmma_f16(a1, b, c[1][t]);
    }
  }

  const int n  = lane & 15;
  const int mh = (lane >> 4) << 3;
#pragma unroll
  for (int mi = 0; mi < 2; ++mi) {
#pragma unroll
    for (int t = 0; t < 4; ++t) {
      const int col = col0 + t * 16 + n;
      const float bj = bias ? bias[col] : 0.0f;
#pragma unroll
      for (int r = 0; r < 8; ++r) {
        const int row = row0 + mi * 16 + mh + r;
        float val = (c[mi][t][r] + bj) * scl;
        if (doRelu) val = fmaxf(val, 0.0f);
        const long idx = (long)row * J + col;
        if (resid) val += resid[idx];
        if (outF) outF[idx] = val;
        if (outH) outH[idx] = (_Float16)val;
        if (outQKV) {
          const int tok = row >> 1, b = row & 1;
          const int h = col / kHD, hd = col % kHD;
          outQKV[(((size_t)(b * kH + h)) * kT + tok) * kHD + hd] = (_Float16)val;
        }
      }
    }
  }
}

// ---------------------------------------------------------------------------
// Flash attention: Q,K,V f16 in [B*H][T][96]. One wave = 16 Q rows, streams K/V
// in 32-column steps with online softmax. Writes f16 attn in [tok][D] layout.
// ---------------------------------------------------------------------------
__global__ __launch_bounds__(256) void flash_attn_k(
    const _Float16* __restrict__ Qh, const _Float16* __restrict__ Kh,
    const _Float16* __restrict__ Vh, _Float16* __restrict__ attnH) {
  __shared__ _Float16 plds[8][16 * 34];
  const int wave = threadIdx.x >> 5, lane = threadIdx.x & 31;
  const int task = blockIdx.x * 8 + wave;   // B*H*(T/16) = 2048 tasks
  const int bh = task >> 7;                 // T/16 = 128 row-blocks per head
  const int m0 = (task & 127) << 4;
  const _Float16* Qb = Qh + (size_t)bh * kT * kHD;
  const _Float16* Kb = Kh + (size_t)bh * kT * kHD;
  const _Float16* Vb = Vh + (size_t)bh * kT * kHD;

  v16h aq[3];
#pragma unroll
  for (int kt = 0; kt < 3; ++kt) aq[kt] = frag_a(Qb, kHD, m0, kt * 32);

  float mrow[8], lrow[8];
  v8f acc[6];
  v8f zero = {};
#pragma unroll
  for (int r = 0; r < 8; ++r) { mrow[r] = -1e30f; lrow[r] = 0.0f; }
#pragma unroll
  for (int nt = 0; nt < 6; ++nt) acc[nt] = zero;

  const int n  = lane & 15;
  const int mh = (lane >> 4) << 3;
  _Float16* pt = plds[wave];

  for (int s0 = 0; s0 < kT; s0 += 32) {
    // S tile (16 x 32): scale already folded into Q
    v8f sc0 = zero, sc1 = zero;
#pragma unroll
    for (int kt = 0; kt < 3; ++kt) {
      v16h bk0 = frag_b_wt(Kb, kHD, s0,      kt * 32);
      v16h bk1 = frag_b_wt(Kb, kHD, s0 + 16, kt * 32);
      sc0 = wmma_f16(aq[kt], bk0, sc0);
      sc1 = wmma_f16(aq[kt], bk1, sc1);
    }

    // online softmax: C layout => lane = N, VGPR r = row (r or r+8 per half);
    // row reductions stay inside each 16-lane half (xor masks < 16)
#pragma unroll
    for (int r = 0; r < 8; ++r) {
      float mx = fmaxf(sc0[r], sc1[r]);
      mx = fmaxf(mx, __shfl_xor(mx, 1, 32));
      mx = fmaxf(mx, __shfl_xor(mx, 2, 32));
      mx = fmaxf(mx, __shfl_xor(mx, 4, 32));
      mx = fmaxf(mx, __shfl_xor(mx, 8, 32));
      const float mnew  = fmaxf(mrow[r], mx);
      const float alpha = __expf(mrow[r] - mnew);
      const float p0 = __expf(sc0[r] - mnew);
      const float p1 = __expf(sc1[r] - mnew);
      sc0[r] = p0; sc1[r] = p1;
      float rs = p0 + p1;
      rs += __shfl_xor(rs, 1, 32);
      rs += __shfl_xor(rs, 2, 32);
      rs += __shfl_xor(rs, 4, 32);
      rs += __shfl_xor(rs, 8, 32);
      lrow[r] = lrow[r] * alpha + rs;
      mrow[r] = mnew;
#pragma unroll
      for (int nt = 0; nt < 6; ++nt) acc[nt][r] *= alpha;
    }

    // P: C-layout -> A-layout via LDS round trip (wave-local tile, stride 34)
#pragma unroll
    for (int r = 0; r < 8; ++r) {
      pt[(mh + r) * 34 + n]      = (_Float16)sc0[r];
      pt[(mh + r) * 34 + 16 + n] = (_Float16)sc1[r];
    }
    __syncthreads();                    // uniform trip count across all 8 waves
    v16h ap = frag_a(pt, 34, 0, 0);
    __syncthreads();

    // O += P @ V : B(k=s_local, n=d) = V[s0+k][d]
#pragma unroll
    for (int nt = 0; nt < 6; ++nt) {
      union { unsigned u[8]; v16h v; } bv;
      const int d  = nt * 16 + n;
      const int kb = s0 + ((lane >> 4) << 4);
#pragma unroll
      for (int vv = 0; vv < 8; ++vv) {
        union { _Float16 h[2]; unsigned u; } pk;
        pk.h[0] = Vb[(size_t)(kb + 2 * vv)     * kHD + d];
        pk.h[1] = Vb[(size_t)(kb + 2 * vv + 1) * kHD + d];
        bv.u[vv] = pk.u;
      }
      acc[nt] = wmma_f16(ap, bv.v, acc[nt]);
    }
  }

  // epilogue: O / l, scatter to token-major [tok][D] f16
  const int b = bh >> 3, h = bh & 7;
#pragma unroll
  for (int r = 0; r < 8; ++r) {
    const int tok = m0 + mh + r;
    const float inv = 1.0f / lrow[r];
#pragma unroll
    for (int nt = 0; nt < 6; ++nt)
      attnH[((size_t)(tok * kB + b)) * kD + h * kHD + nt * 16 + n] =
          (_Float16)(acc[nt][r] * inv);
  }
}

// ---------------------------------------------------------------------------
// LayerNorm: one wave per row (D=768 => 24 elems/lane), shuffle reductions
// ---------------------------------------------------------------------------
__global__ __launch_bounds__(256) void layernorm_k(
    const float* __restrict__ X, const float* __restrict__ g, const float* __restrict__ b,
    float* __restrict__ outF, _Float16* __restrict__ outH, int D) {
  const int wave = threadIdx.x >> 5, lane = threadIdx.x & 31;
  const long row = (long)blockIdx.x * 8 + wave;
  const float* x = X + row * D;
  float s = 0.0f, s2 = 0.0f;
  for (int i = lane; i < D; i += 32) { float v = x[i]; s += v; s2 += v * v; }
#pragma unroll
  for (int off = 16; off > 0; off >>= 1) {
    s  += __shfl_xor(s,  off, 32);
    s2 += __shfl_xor(s2, off, 32);
  }
  const float mu = s / D;
  const float var = s2 / D - mu * mu;
  const float rstd = rsqrtf(var + 1e-5f);
  for (int i = lane; i < D; i += 32) {
    float v = (x[i] - mu) * rstd * g[i] + b[i];
    if (outF) outF[row * D + i] = v;
    if (outH) outH[row * D + i] = (_Float16)v;
  }
}

// ---------------------------------------------------------------------------
extern "C" void kernel_launch(void* const* d_in, const int* in_sizes, int n_in,
                              void* d_out, int out_size, void* d_ws, size_t ws_size,
                              hipStream_t stream) {
  (void)in_sizes; (void)n_in; (void)out_size; (void)ws_size;
  const float* x     = (const float*)d_in[0];
  const float* wq    = (const float*)d_in[1];
  const float* bq    = (const float*)d_in[2];
  const float* wk    = (const float*)d_in[3];
  const float* bk    = (const float*)d_in[4];
  const float* wv    = (const float*)d_in[5];
  const float* bv    = (const float*)d_in[6];
  const float* wo    = (const float*)d_in[7];
  const float* bo    = (const float*)d_in[8];
  const float* ln1_g = (const float*)d_in[9];
  const float* ln1_b = (const float*)d_in[10];
  const float* w1    = (const float*)d_in[11];
  const float* b1    = (const float*)d_in[12];
  const float* w2    = (const float*)d_in[13];
  const float* b2    = (const float*)d_in[14];
  const float* ln2_g = (const float*)d_in[15];
  const float* ln2_b = (const float*)d_in[16];

  size_t off = 0;
  auto take = [&](size_t bytes) -> void* {
    void* p = (char*)d_ws + off;
    off += (bytes + 255) & ~(size_t)255;
    return p;
  };
  _Float16* xh    = (_Float16*)take((size_t)kN * kD * 2);
  _Float16* wqh   = (_Float16*)take((size_t)kD * kD * 2);
  _Float16* wkh   = (_Float16*)take((size_t)kD * kD * 2);
  _Float16* wvh   = (_Float16*)take((size_t)kD * kD * 2);
  _Float16* woh   = (_Float16*)take((size_t)kD * kD * 2);
  _Float16* w1h   = (_Float16*)take((size_t)kFFN * kD * 2);
  _Float16* w2h   = (_Float16*)take((size_t)kD * kFFN * 2);
  _Float16* qh    = (_Float16*)take((size_t)kN * kD * 2);
  _Float16* kh    = (_Float16*)take((size_t)kN * kD * 2);
  _Float16* vh    = (_Float16*)take((size_t)kN * kD * 2);
  _Float16* atth  = (_Float16*)take((size_t)kN * kD * 2);
  float*    y1    = (float*)take((size_t)kN * kD * 4);
  float*    ln1F  = (float*)take((size_t)kN * kD * 4);
  _Float16* ln1H  = (_Float16*)take((size_t)kN * kD * 2);
  _Float16* hh    = (_Float16*)take((size_t)kN * kFFN * 2);
  float*    y2    = (float*)take((size_t)kN * kD * 4);

  auto cvt = [&](const float* src, _Float16* dst, long nelem) {
    long blocks = (nelem + 255) / 256;
    if (blocks > 4096) blocks = 4096;
    f32_to_f16_k<<<(int)blocks, 256, 0, stream>>>(src, dst, nelem);
  };
  cvt(x,  xh,  (long)kN * kD);
  cvt(wq, wqh, (long)kD * kD);
  cvt(wk, wkh, (long)kD * kD);
  cvt(wv, wvh, (long)kD * kD);
  cvt(wo, woh, (long)kD * kD);
  cvt(w1, w1h, (long)kFFN * kD);
  cvt(w2, w2h, (long)kD * kFFN);

  auto gemm = [&](const _Float16* A, const _Float16* W, const float* bias,
                  const float* resid, float* oF, _Float16* oH, _Float16* oQKV,
                  int N, int J, int K, float scl, int relu) {
    int blocks = (N / 256) * (J / 64);
    gemm_wt_k<<<blocks, 256, 0, stream>>>(A, W, bias, resid, oF, oH, oQKV,
                                          N, J, K, scl, relu);
  };

  const float qscale = 0.1020620726159658f;  // 1/sqrt(96), folded into Q
  gemm(xh, wqh, bq, nullptr, nullptr, nullptr, qh, kN, kD, kD, qscale, 0);
  gemm(xh, wkh, bk, nullptr, nullptr, nullptr, kh, kN, kD, kD, 1.0f, 0);
  gemm(xh, wvh, bv, nullptr, nullptr, nullptr, vh, kN, kD, kD, 1.0f, 0);

  flash_attn_k<<<(kB * kH * (kT / 16)) / 8, 256, 0, stream>>>(qh, kh, vh, atth);

  gemm(atth, woh, bo, x, y1, nullptr, nullptr, kN, kD, kD, 1.0f, 0);
  layernorm_k<<<kN / 8, 256, 0, stream>>>(y1, ln1_g, ln1_b, ln1F, ln1H, kD);

  gemm(ln1H, w1h, b1, nullptr, nullptr, hh, nullptr, kN, kFFN, kD, 1.0f, 1);
  gemm(hh, w2h, b2, ln1F, y2, nullptr, nullptr, kN, kD, kFFN, 1.0f, 0);
  layernorm_k<<<kN / 8, 256, 0, stream>>>(y2, ln2_g, ln2_b, (float*)d_out, nullptr, kD);
}
